// MultiheadAttention_12163347382558
// MI455X (gfx1250) — matrix operand order repaired and checked by reading, not compiled, not run on hardware
//
#include <hip/hip_runtime.h>
#include <hip/hip_bf16.h>
#include <math.h>

// ---------------------------------------------------------------------------
// MHA forward for MI455X (gfx1250), wave32, WMMA f32_16x16x32_f16.
// B=4, S=2048, D_IN=D=1024, H=16, HD=64.
// d_out = out[B,S,D] (f32) ++ attn[B,H,S,S] (f32).
// ---------------------------------------------------------------------------

typedef __attribute__((ext_vector_type(16))) _Float16 v16h;
typedef __attribute__((ext_vector_type(8)))  _Float16 v8h;
typedef __attribute__((ext_vector_type(4)))  _Float16 v4h;
typedef __attribute__((ext_vector_type(8)))  float    v8f;
typedef __attribute__((ext_vector_type(4)))  float    v4f;

#define WMMA_F16(a, b, c) \
  __builtin_amdgcn_wmma_f32_16x16x32_f16(false, (a), false, (b), (short)0, (c), false, false)

constexpr int BB  = 4;
constexpr int SS  = 2048;
constexpr int DIN = 1024;
constexpr int DD  = 1024;
constexpr int HH  = 16;
constexpr int HD  = 64;

constexpr int LDK = 40;  // padded LDS row stride in f16 (80 B, 16B-aligned rows)

union FragU { v16h v; v8h h[2]; };

// ---------------------------------------------------------------------------
// Weight f32 -> f16 conversion (vector-of-4 per thread).
// ---------------------------------------------------------------------------
__global__ __launch_bounds__(256) void cvt_f16_kernel(const float* __restrict__ src,
                                                      _Float16* __restrict__ dst, int n) {
  int i = (blockIdx.x * 256 + threadIdx.x) * 4;
  if (i < n) {
    v4f v = *(const v4f*)(src + i);
    v4h h;
    h.x = (_Float16)v.x; h.y = (_Float16)v.y; h.z = (_Float16)v.z; h.w = (_Float16)v.w;
    *(v4h*)(dst + i) = h;
  }
}

// ---------------------------------------------------------------------------
// GEMM: out(m,n) = sum_k A[m,k] * W[n,k] + bias[n]
//   AF32  : A is f32 (on-the-fly f16 convert) else A is f16.
//   MODE 0: f16 output scattered to QKV layout [B,H,S,HD]   (m=(b,s), n=(h,hd))
//   MODE 1: f32 output flat [M,N]
// Block 128x128, 256 threads = 8 waves (2 M x 4 N), wave tile 64x32 (4x2 WMMA).
// ---------------------------------------------------------------------------
template <bool AF32, int MODE>
__global__ __launch_bounds__(256) void gemm_kernel(
    const float* __restrict__ Af32, const _Float16* __restrict__ Af16,
    const _Float16* __restrict__ Wt, const float* __restrict__ bias,
    _Float16* __restrict__ out16, float* __restrict__ outf32,
    int M, int N, int K) {
  __shared__ __align__(16) _Float16 As[128 * LDK];
  __shared__ __align__(16) _Float16 Bs[128 * LDK];

  const int tid = threadIdx.x;
  const int lane = tid & 31, wave = tid >> 5;
  const int l16 = lane & 15, laneHalf = lane >> 4;
  const int wm = wave >> 2, wn = wave & 3;
  const int m0 = blockIdx.y * 128;
  const int n0 = blockIdx.x * 128;
  const int abase = laneHalf ? 8 : 0;   // A-frag K chunk base
  const int kbase = laneHalf ? 8 : 0;   // B-frag K base

  v8f acc[4][2] = {};

  for (int k0 = 0; k0 < K; k0 += 32) {
    // prefetch next K tile (lowers to global_prefetch_b8)
    if (k0 + 32 < K) {
      if constexpr (AF32) {
        __builtin_prefetch(Af32 + (size_t)(m0 + (tid >> 1)) * K + k0 + 32, 0, 1);
      } else {
        __builtin_prefetch(Af16 + (size_t)(m0 + (tid >> 1)) * K + k0 + 32, 0, 1);
      }
      __builtin_prefetch(Wt + (size_t)(n0 + (tid >> 1)) * K + k0 + 32, 0, 1);
    }

    // ---- stage A tile (128 x 32) into LDS as f16 ----
    if constexpr (AF32) {
#pragma unroll
      for (int it = 0; it < 4; ++it) {
        int chunk = tid + it * 256;                 // 1024 chunks of 4 floats
        int r = chunk >> 3, c4 = (chunk & 7) * 4;
        v4f v = *(const v4f*)(Af32 + (size_t)(m0 + r) * K + k0 + c4);
        v4h h;
        h.x = (_Float16)v.x; h.y = (_Float16)v.y; h.z = (_Float16)v.z; h.w = (_Float16)v.w;
        *(v4h*)&As[r * LDK + c4] = h;
      }
    } else {
#pragma unroll
      for (int it = 0; it < 2; ++it) {
        int chunk = tid + it * 256;                 // 512 chunks of 8 f16
        int r = chunk >> 2, c8 = (chunk & 3) * 8;
        *(v8h*)&As[r * LDK + c8] = *(const v8h*)(Af16 + (size_t)(m0 + r) * K + k0 + c8);
      }
    }
    // ---- stage W tile (128 rows of n, 32 k) ----
#pragma unroll
    for (int it = 0; it < 2; ++it) {
      int chunk = tid + it * 256;
      int r = chunk >> 2, c8 = (chunk & 3) * 8;
      *(v8h*)&Bs[r * LDK + c8] = *(const v8h*)(Wt + (size_t)(n0 + r) * K + k0 + c8);
    }
    __syncthreads();

    // ---- fragments ----
    v16h afr[4];
#pragma unroll
    for (int sm = 0; sm < 4; ++sm) {
      int row = wm * 64 + sm * 16 + l16;
      FragU u;
      u.h[0] = *(const v8h*)&As[row * LDK + abase];
      u.h[1] = *(const v8h*)&As[row * LDK + abase + 16];
      afr[sm] = u.v;
    }
    v16h bfr[2];
#pragma unroll
    for (int sn = 0; sn < 2; ++sn) {
      int row = wn * 32 + sn * 16 + l16;
      FragU u;
      u.h[0] = *(const v8h*)&Bs[row * LDK + kbase];
      u.h[1] = *(const v8h*)&Bs[row * LDK + kbase + 16];
      bfr[sn] = u.v;
    }
#pragma unroll
    for (int sm = 0; sm < 4; ++sm)
#pragma unroll
      for (int sn = 0; sn < 2; ++sn)
        acc[sm][sn] = WMMA_F16(afr[sm], bfr[sn], acc[sm][sn]);
    __syncthreads();
  }

  // ---- epilogue: bias + store ----
#pragma unroll
  for (int sm = 0; sm < 4; ++sm) {
#pragma unroll
    for (int sn = 0; sn < 2; ++sn) {
      int n = n0 + wn * 32 + sn * 16 + l16;
      float bv = bias[n];
#pragma unroll
      for (int i = 0; i < 8; ++i) {
        int m = m0 + wm * 64 + sm * 16 + i + laneHalf * 8;
        float val = acc[sm][sn][i] + bv;
        if constexpr (MODE == 0) {
          int b = m >> 11, s = m & (SS - 1);
          int h = n >> 6, hd = n & (HD - 1);
          out16[(((size_t)(b * HH + h) * SS + s) << 6) + hd] = (_Float16)val;
        } else {
          outf32[(size_t)m * N + n] = val;
        }
      }
    }
  }
}

// ---------------------------------------------------------------------------
// V [BH,S,HD] f16 -> Vt [BH,HD,S] f16 (32x32 LDS tile transpose)
// ---------------------------------------------------------------------------
__global__ __launch_bounds__(256) void transpose_v_kernel(const _Float16* __restrict__ V16,
                                                          _Float16* __restrict__ Vt) {
  __shared__ _Float16 tile[32][34];
  int blk = blockIdx.x;
  int s0 = (blk & 63) * 32;
  int hd0 = ((blk >> 6) & 1) * 32;
  int bh = blk >> 7;
  const _Float16* src = V16 + (size_t)bh * SS * HD;
  _Float16* dst = Vt + (size_t)bh * HD * SS;
  int tid = threadIdx.x;
#pragma unroll
  for (int i = 0; i < 4; ++i) {
    int e = tid + i * 256;
    int r = e >> 5, c = e & 31;
    tile[r][c] = src[(size_t)(s0 + r) * HD + hd0 + c];
  }
  __syncthreads();
#pragma unroll
  for (int i = 0; i < 4; ++i) {
    int e = tid + i * 256;
    int r = e >> 5, c = e & 31;
    dst[(size_t)(hd0 + r) * SS + s0 + c] = tile[c][r];
  }
}

// ---------------------------------------------------------------------------
// Attention: per block = (b, h, 128-query rows); 8 waves x 16 rows each.
// Pass 1: online row max / sum-exp over 64-key blocks (WMMA QK^T); per-lane
//         partial reduction across the 4 subtiles, then ONE cross-lane
//         (ds_bpermute) reduction per row per 64 keys.
// Pass 2: recompute logits, write attn probs (f32), stage P via LDS, PV WMMA.
// ---------------------------------------------------------------------------
__global__ __launch_bounds__(256) void attn_kernel(
    const _Float16* __restrict__ Q16, const _Float16* __restrict__ K16,
    const _Float16* __restrict__ Vt16, const unsigned char* __restrict__ mask,
    float* __restrict__ attn_out, _Float16* __restrict__ vals16) {
  __shared__ __align__(16) _Float16 pst[8][16 * 32];  // per-wave P staging

  const int tid = threadIdx.x;
  const int lane = tid & 31, wave = tid >> 5;
  const int l16 = lane & 15, laneHalf = lane >> 4;
  const int qblk = blockIdx.x & 15;   // S/128 = 16
  const int bh = blockIdx.x >> 4;     // b*H + h
  const int b = bh >> 4;              // H = 16
  const int hq = bh & 15;
  const int q0 = qblk * 128 + wave * 16;

  const _Float16* Qb = Q16 + (size_t)bh * SS * HD;
  const _Float16* Kb = K16 + (size_t)bh * SS * HD;
  const _Float16* Vb = Vt16 + (size_t)bh * HD * SS;
  const unsigned char* mb = mask + (size_t)b * SS;
  float* attnb = attn_out + (size_t)bh * SS * SS;

  const int abase = laneHalf ? 8 : 0;
  const int kbase = laneHalf ? 8 : 0;
  const float scale = 0.125f;  // 1/sqrt(64)

  // Q fragments held in registers for both passes (hd 0..31, 32..63)
  v16h qfr[2];
  {
    const _Float16* qr = Qb + (size_t)(q0 + l16) * HD;
#pragma unroll
    for (int f = 0; f < 2; ++f) {
      FragU u;
      u.h[0] = *(const v8h*)(qr + f * 32 + abase);
      u.h[1] = *(const v8h*)(qr + f * 32 + abase + 16);
      qfr[f] = u.v;
    }
  }

  float mrow[8], srow[8];
#pragma unroll
  for (int i = 0; i < 8; ++i) { mrow[i] = -INFINITY; srow[i] = 0.0f; }

  // ----- pass 1: online softmax stats over 64-key blocks -----
  for (int kb = 0; kb < SS; kb += 64) {
    v8f cs[4];
#pragma unroll
    for (int sub = 0; sub < 4; ++sub) {
      int ks = kb + sub * 16;
      const _Float16* kr = Kb + (size_t)(ks + l16) * HD;
      v8f c = {};
#pragma unroll
      for (int f = 0; f < 2; ++f) {
        FragU u;
        u.h[0] = *(const v8h*)(kr + f * 32 + kbase);
        u.h[1] = *(const v8h*)(kr + f * 32 + kbase + 16);
        c = WMMA_F16(qfr[f], u.v, c);
      }
      bool masked = mb[ks + l16] != 0;
#pragma unroll
      for (int i = 0; i < 8; ++i) {
        float l = c[i] * scale;
        c[i] = masked ? -9e15f : l;
      }
      cs[sub] = c;
    }
#pragma unroll
    for (int i = 0; i < 8; ++i) {
      // per-lane partial max across the 4 subtiles, then one lane reduction
      float pm = fmaxf(fmaxf(cs[0][i], cs[1][i]), fmaxf(cs[2][i], cs[3][i]));
      pm = fmaxf(pm, __shfl_xor(pm, 1, 32));
      pm = fmaxf(pm, __shfl_xor(pm, 2, 32));
      pm = fmaxf(pm, __shfl_xor(pm, 4, 32));
      pm = fmaxf(pm, __shfl_xor(pm, 8, 32));
      float mn = fmaxf(mrow[i], pm);
      float t = __expf(cs[0][i] - mn) + __expf(cs[1][i] - mn) +
                __expf(cs[2][i] - mn) + __expf(cs[3][i] - mn);
      t += __shfl_xor(t, 1, 32);
      t += __shfl_xor(t, 2, 32);
      t += __shfl_xor(t, 4, 32);
      t += __shfl_xor(t, 8, 32);
      srow[i] = srow[i] * __expf(mrow[i] - mn) + t;
      mrow[i] = mn;
    }
  }
  float inv[8];
#pragma unroll
  for (int i = 0; i < 8; ++i) inv[i] = 1.0f / srow[i];

  // ----- pass 2: attn write + PV accumulation -----
  v8f oacc[4] = {};
  _Float16* ps = &pst[wave][0];

  for (int kb = 0; kb < SS; kb += 32) {
#pragma unroll
    for (int sub = 0; sub < 2; ++sub) {
      int ks = kb + sub * 16;
      const _Float16* kr = Kb + (size_t)(ks + l16) * HD;
      v8f c = {};
#pragma unroll
      for (int f = 0; f < 2; ++f) {
        FragU u;
        u.h[0] = *(const v8h*)(kr + f * 32 + kbase);
        u.h[1] = *(const v8h*)(kr + f * 32 + kbase + 16);
        c = WMMA_F16(qfr[f], u.v, c);
      }
      int col = ks + l16;
      bool masked = mb[col] != 0;
#pragma unroll
      for (int i = 0; i < 8; ++i) {
        float l = c[i] * scale;
        l = masked ? -9e15f : l;
        float p = __expf(l - mrow[i]) * inv[i];
        int row = i + laneHalf * 8;
        attnb[(size_t)(q0 + row) * SS + col] = p;       // attention probabilities
        ps[row * 32 + sub * 16 + l16] = (_Float16)p;    // stage for PV
      }
    }
    // wave-private LDS: order the cross-lane store->load explicitly
    asm volatile("s_wait_dscnt 0" ::: "memory");

    FragU up;
    up.h[0] = *(const v8h*)&ps[l16 * 32 + abase];
    up.h[1] = *(const v8h*)&ps[l16 * 32 + abase + 16];

#pragma unroll
    for (int sn = 0; sn < 4; ++sn) {
      const _Float16* vr = Vb + (size_t)(sn * 16 + l16) * SS + kb;
      FragU uv;
      uv.h[0] = *(const v8h*)(vr + kbase);
      uv.h[1] = *(const v8h*)(vr + kbase + 16);
      oacc[sn] = WMMA_F16(up.v, uv.v, oacc[sn]);
    }
    asm volatile("s_wait_dscnt 0" ::: "memory");
  }

  // store O into vals16[b, s, h*HD + hd] (f16 for the output projection)
#pragma unroll
  for (int sn = 0; sn < 4; ++sn) {
#pragma unroll
    for (int i = 0; i < 8; ++i) {
      int row = q0 + i + laneHalf * 8;
      int hd = sn * 16 + l16;
      vals16[(size_t)(b * SS + row) * DD + hq * HD + hd] = (_Float16)oacc[sn][i];
    }
  }
}

// ---------------------------------------------------------------------------
// Launch
// ---------------------------------------------------------------------------
extern "C" void kernel_launch(void* const* d_in, const int* in_sizes, int n_in,
                              void* d_out, int out_size, void* d_ws, size_t ws_size,
                              hipStream_t stream) {
  const float* in_key   = (const float*)d_in[0];
  const float* in_query = (const float*)d_in[1];
  const float* in_value = (const float*)d_in[2];
  const unsigned char* mask = (const unsigned char*)d_in[3];
  const float* Wq = (const float*)d_in[4];  const float* bq = (const float*)d_in[5];
  const float* Wk = (const float*)d_in[6];  const float* bk = (const float*)d_in[7];
  const float* Wv = (const float*)d_in[8];  const float* bv = (const float*)d_in[9];
  const float* Wo = (const float*)d_in[10]; const float* bo = (const float*)d_in[11];

  float* out  = (float*)d_out;
  float* attn = out + (size_t)BB * SS * DD;

  // workspace layout (f16 elements)
  _Float16* ws16   = (_Float16*)d_ws;
  _Float16* Wq16   = ws16;                                   // 1M
  _Float16* Wk16   = Wq16 + (size_t)DD * DIN;                // 1M
  _Float16* Wv16   = Wk16 + (size_t)DD * DIN;                // 1M
  _Float16* Wo16   = Wv16 + (size_t)DD * DIN;                // 1M
  _Float16* Q16    = Wo16 + (size_t)DD * DD;                 // 8M
  _Float16* K16    = Q16 + (size_t)BB * HH * SS * HD;        // 8M
  _Float16* V16    = K16 + (size_t)BB * HH * SS * HD;        // 8M
  _Float16* Vt16   = V16 + (size_t)BB * HH * SS * HD;        // 8M
  _Float16* vals16 = Vt16 + (size_t)BB * HH * HD * SS;       // 8M

  const int M = BB * SS;          // 8192
  const int nW = DD * DIN;        // 1048576

  cvt_f16_kernel<<<nW / 1024, 256, 0, stream>>>(Wq, Wq16, nW);
  cvt_f16_kernel<<<nW / 1024, 256, 0, stream>>>(Wk, Wk16, nW);
  cvt_f16_kernel<<<nW / 1024, 256, 0, stream>>>(Wv, Wv16, nW);
  cvt_f16_kernel<<<nW / 1024, 256, 0, stream>>>(Wo, Wo16, nW);

  dim3 gblk(DD / 128, M / 128);   // (8, 64)
  gemm_kernel<true, 0><<<gblk, 256, 0, stream>>>(in_query, nullptr, Wq16, bq, Q16, nullptr,
                                                 M, DD, DIN);
  gemm_kernel<true, 0><<<gblk, 256, 0, stream>>>(in_key, nullptr, Wk16, bk, K16, nullptr,
                                                 M, DD, DIN);
  gemm_kernel<true, 0><<<gblk, 256, 0, stream>>>(in_value, nullptr, Wv16, bv, V16, nullptr,
                                                 M, DD, DIN);

  transpose_v_kernel<<<BB * HH * (SS / 32) * (HD / 32), 256, 0, stream>>>(V16, Vt16);

  attn_kernel<<<BB * HH * (SS / 128), 256, 0, stream>>>(Q16, K16, Vt16, mask, attn, vals16);

  gemm_kernel<false, 1><<<gblk, 256, 0, stream>>>(nullptr, vals16, Wo16, bo, nullptr, out,
                                                  M, DD, DD);
}
